// NaViT_52115133170417
// MI455X (gfx1250) — compile-verified, matrix operand order checked by reading, not run on hardware
//
#include <hip/hip_runtime.h>
#include <hip/hip_bf16.h>
#include <math.h>

// ---------------- model dims ----------------
#define BB    4
#define NN    1024
#define DIM   768
#define HEADS 12
#define DHEAD 64
#define INNER 768
#define MLP   3072
#define DEPTH 4
#define PDIM  768
#define GRID  16
#define MTOK  (BB*NN)          // 4096 packed tokens

typedef __attribute__((ext_vector_type(16))) __bf16 v16bf;
typedef __attribute__((ext_vector_type(8)))  float  v8f;
typedef __attribute__((ext_vector_type(4)))  unsigned int v4u;

union Frag16 { v16bf h; v4u u[2]; unsigned short s[16]; };
union AccF   { v8f f; float e[8]; };
union Pack8  { unsigned short s[8]; v4u v; };

static __device__ inline unsigned short f2bf(float f) {
    unsigned int u = __float_as_uint(f);
    u = (u + 0x7FFFu + ((u >> 16) & 1u)) >> 16;
    return (unsigned short)u;
}
static __device__ inline float bf2f(unsigned short s) {
    return __uint_as_float(((unsigned int)s) << 16);
}
static __device__ inline v8f wmma_bf16(const Frag16& a, const Frag16& b, v8f c) {
    return __builtin_amdgcn_wmma_f32_16x16x32_bf16(false, a.h, false, b.h,
                                                   (short)0, c, false, false);
}

// =====================================================================
// PatchNorm (eval) + factorized pos embed -> bf16 activation [MTOK, PDIM]
// =====================================================================
__global__ __launch_bounds__(256) void patchnorm_kernel(
    const float* __restrict__ patches, const int* __restrict__ pos_h,
    const int* __restrict__ pos_w, const unsigned char* __restrict__ pad,
    const float* __restrict__ pn_n, const float* __restrict__ pn_mean,
    const float* __restrict__ pn_m2, const float* __restrict__ peh,
    const float* __restrict__ pew, unsigned short* __restrict__ out)
{
    int tok = blockIdx.x;                       // 0..4095
    int ph = pos_h[tok], pw = pos_w[tok];
    float nval = pn_n[ph * GRID + pw];
    float clip = fmaxf(nval, 1.0f);
    bool zn = (nval == 0.0f);
    bool isPad = pad[tok] != 0;
    size_t cellOff = (size_t)(ph * GRID + pw) * PDIM;
    for (int c = threadIdx.x; c < PDIM; c += 256) {
        float var = zn ? 1.0f : pn_m2[cellOff + c] / clip;
        float sd = sqrtf(var);
        float xv = (patches[(size_t)tok * PDIM + c] - pn_mean[cellOff + c]) / (sd + 0.01f);
        if (isPad) xv = 0.0f;
        xv += peh[(size_t)ph * PDIM + c] + pew[(size_t)pw * PDIM + c];
        out[(size_t)tok * PDIM + c] = f2bf(xv);
    }
}

// =====================================================================
// LayerNorm over last dim (768). One block per token.
// =====================================================================
template<bool OUT_BF16>
__global__ __launch_bounds__(256) void ln_kernel(
    const float* __restrict__ in, const float* __restrict__ g,
    float* __restrict__ outF, unsigned short* __restrict__ outB)
{
    __shared__ float red[256];
    int tok = blockIdx.x, t = threadIdx.x;
    const float* row = in + (size_t)tok * DIM;
    float v[3];
    float s = 0.f;
    #pragma unroll
    for (int i = 0; i < 3; i++) { v[i] = row[t + 256 * i]; s += v[i]; }
    red[t] = s; __syncthreads();
    for (int st = 128; st > 0; st >>= 1) { if (t < st) red[t] += red[t + st]; __syncthreads(); }
    float mean = red[0] / (float)DIM; __syncthreads();
    float s2 = 0.f;
    #pragma unroll
    for (int i = 0; i < 3; i++) { float d = v[i] - mean; s2 += d * d; }
    red[t] = s2; __syncthreads();
    for (int st = 128; st > 0; st >>= 1) { if (t < st) red[t] += red[t + st]; __syncthreads(); }
    float rstd = rsqrtf(red[0] / (float)DIM + 1e-5f);
    #pragma unroll
    for (int i = 0; i < 3; i++) {
        int c = t + 256 * i;
        float o = (v[i] - mean) * rstd * g[c];
        if (OUT_BF16) outB[(size_t)tok * DIM + c] = f2bf(o);
        else          outF[(size_t)tok * DIM + c] = o;
    }
}

// =====================================================================
// Generic GEMM:  out[M,N] = A_bf16[M,K] x W_f32[K,N] (+bias)(+res)(+gelu)
// block = 256 thr = 8 waves, 128x128 block tile, 32x64 per-wave tile,
// K step 32 -> 8x v_wmma_f32_16x16x32_bf16 per wave per K step.
// =====================================================================
template<bool OUT_BF16, bool GELU, bool RES>
__global__ __launch_bounds__(256) void gemm_kernel(
    const unsigned short* __restrict__ A, const float* __restrict__ W,
    const float* __restrict__ bias, float* __restrict__ outF,
    unsigned short* __restrict__ outB, const float* __restrict__ res,
    int N, int K)
{
    __shared__ __align__(16) unsigned short Alds[128 * 32];   // [m][k]
    __shared__ __align__(16) unsigned short Wlds[128 * 32];   // transposed: [n][k]
    int t = threadIdx.x, lane = t & 31, wid = t >> 5;
    int bm = blockIdx.y * 128, bn = blockIdx.x * 128;
    int mo = (wid >> 1) * 32;           // wave M offset in tile (0,32,64,96)
    int no = (wid & 1) * 64;            // wave N offset in tile (0,64)
    int r  = lane & 15;
    int kb = (lane >> 4) * 8;           // documented wave32 A/B K packing
    v8f acc[2][4] = {};

    int arow = t >> 1, aseg = (t & 1) * 16;          // A staging: 2x16B per thread
    int wn = t & 127, wkg = (t >> 7) * 16;           // W staging: 16 k-rows per thread
    for (int k0 = 0; k0 < K; k0 += 32) {
        __syncthreads();
        // ---- stage A tile (128 x 32 bf16), contiguous b128 loads
        {
            const unsigned short* ap = A + (size_t)(bm + arow) * K + k0 + aseg;
            *(v4u*)&Alds[arow * 32 + aseg]     = *(const v4u*)ap;
            *(v4u*)&Alds[arow * 32 + aseg + 8] = *(const v4u*)(ap + 8);
        }
        // ---- stage W tile transposed, f32 -> bf16
        {
            Pack8 p0, p1;
            const float* wp = W + (size_t)(k0 + wkg) * N + bn + wn;
            #pragma unroll
            for (int i = 0; i < 8; i++) p0.s[i] = f2bf(wp[(size_t)i * N]);
            #pragma unroll
            for (int i = 0; i < 8; i++) p1.s[i] = f2bf(wp[(size_t)(i + 8) * N]);
            *(v4u*)&Wlds[wn * 32 + wkg]     = p0.v;
            *(v4u*)&Wlds[wn * 32 + wkg + 8] = p1.v;
        }
        if (k0 + 32 < K)
            __builtin_prefetch(A + (size_t)(bm + arow) * K + k0 + 32 + aseg, 0, 3);
        __syncthreads();
        // ---- fragments (16B ds loads, layout matches ISA VGPR packing)
        Frag16 af[2], bf[4];
        #pragma unroll
        for (int i = 0; i < 2; i++) {
            af[i].u[0] = *(const v4u*)&Alds[(mo + i * 16 + r) * 32 + kb];
            af[i].u[1] = *(const v4u*)&Alds[(mo + i * 16 + r) * 32 + kb + 16];
        }
        #pragma unroll
        for (int tn = 0; tn < 4; tn++) {
            bf[tn].u[0] = *(const v4u*)&Wlds[(no + tn * 16 + r) * 32 + kb];
            bf[tn].u[1] = *(const v4u*)&Wlds[(no + tn * 16 + r) * 32 + kb + 16];
        }
        #pragma unroll
        for (int i = 0; i < 2; i++)
            #pragma unroll
            for (int tn = 0; tn < 4; tn++)
                acc[i][tn] = wmma_bf16(af[i], bf[tn], acc[i][tn]);
    }
    // ---- epilogue (C layout: lane holds (m0+i, c), m0 = (lane>=16)*8)
    int c = lane & 15, m0 = (lane >> 4) * 8;
    #pragma unroll
    for (int im = 0; im < 2; im++) {
        #pragma unroll
        for (int tn = 0; tn < 4; tn++) {
            AccF ac; ac.f = acc[im][tn];
            int col = bn + no + tn * 16 + c;
            float bv = bias ? bias[col] : 0.0f;
            #pragma unroll
            for (int i = 0; i < 8; i++) {
                size_t idx = (size_t)(bm + mo + im * 16 + m0 + i) * N + col;
                float v = ac.e[i] + bv;
                if (GELU) v = 0.5f * v * (1.0f + erff(v * 0.70710678118654752f));
                if (RES)  v += res[idx];
                if (OUT_BF16) outB[idx] = f2bf(v);
                else          outF[idx] = v;
            }
        }
    }
}

// =====================================================================
// QK RMSNorm + V transpose.  One wave per (b,h,n) unit.
// q,k -> bf16 [bh][n][64];  v -> bf16 transposed [bh][64][n]
// =====================================================================
__global__ __launch_bounds__(256) void qkv_transform_kernel(
    const unsigned short* __restrict__ qraw,   // [MTOK, 768]
    const unsigned short* __restrict__ kvraw,  // [MTOK, 1536]  (k | v)
    const float* __restrict__ qg, const float* __restrict__ kg,
    unsigned short* __restrict__ q, unsigned short* __restrict__ k,
    unsigned short* __restrict__ vt)
{
    int lane = threadIdx.x & 31, wid = threadIdx.x >> 5;
    int u = blockIdx.x * 8 + wid;          // 0 .. B*H*N-1
    int n = u & (NN - 1);
    int bh = u >> 10;                      // b*HEADS + h
    int hh = bh % HEADS;
    int b  = bh / HEADS;
    size_t row = (size_t)b * NN + n;
    int d0 = lane * 2;

    // ---- Q rmsnorm
    unsigned int pq = *(const unsigned int*)&qraw[row * INNER + hh * DHEAD + d0];
    float q0 = bf2f((unsigned short)(pq & 0xFFFF));
    float q1 = bf2f((unsigned short)(pq >> 16));
    float ss = q0 * q0 + q1 * q1;
    for (int d = 1; d < 32; d <<= 1) ss += __shfl_xor(ss, d, 32);
    float sc = 8.0f / fmaxf(sqrtf(ss), 1e-12f);        // sqrt(DHEAD)=8
    q0 *= sc * qg[hh * DHEAD + d0]; q1 *= sc * qg[hh * DHEAD + d0 + 1];
    unsigned int po = (unsigned int)f2bf(q0) | ((unsigned int)f2bf(q1) << 16);
    *(unsigned int*)&q[((size_t)bh * NN + n) * DHEAD + d0] = po;

    // ---- K rmsnorm
    unsigned int pk = *(const unsigned int*)&kvraw[row * (2 * INNER) + hh * DHEAD + d0];
    float k0 = bf2f((unsigned short)(pk & 0xFFFF));
    float k1 = bf2f((unsigned short)(pk >> 16));
    float sk = k0 * k0 + k1 * k1;
    for (int d = 1; d < 32; d <<= 1) sk += __shfl_xor(sk, d, 32);
    float sck = 8.0f / fmaxf(sqrtf(sk), 1e-12f);
    k0 *= sck * kg[hh * DHEAD + d0]; k1 *= sck * kg[hh * DHEAD + d0 + 1];
    po = (unsigned int)f2bf(k0) | ((unsigned int)f2bf(k1) << 16);
    *(unsigned int*)&k[((size_t)bh * NN + n) * DHEAD + d0] = po;

    // ---- V transpose (no norm)
    unsigned int pv = *(const unsigned int*)&kvraw[row * (2 * INNER) + INNER + hh * DHEAD + d0];
    vt[((size_t)bh * DHEAD + d0)     * NN + n] = (unsigned short)(pv & 0xFFFF);
    vt[((size_t)bh * DHEAD + d0 + 1) * NN + n] = (unsigned short)(pv >> 16);
}

// =====================================================================
// Flash attention, wave32 WMMA. One wave per 16-row Q block; 32-key chunks.
// q,k: [bh][n][64] bf16, vt: [bh][64][n] bf16, out o: [MTOK, INNER] bf16
// =====================================================================
__global__ __launch_bounds__(256) void attn_kernel(
    const unsigned short* __restrict__ q, const unsigned short* __restrict__ k,
    const unsigned short* __restrict__ vt, const int* __restrict__ image_ids,
    const unsigned char* __restrict__ pad, unsigned short* __restrict__ o)
{
    __shared__ __align__(16) unsigned short Plds[8 * 16 * 32];   // per-wave P tile
    int lane = threadIdx.x & 31, wid = threadIdx.x >> 5;
    int u = blockIdx.x * 8 + wid;            // 0 .. B*H*64-1
    int qb = u & 63;
    int bh = u >> 6;
    int b  = bh / HEADS, hh = bh % HEADS;

    const unsigned short* qm = q  + (size_t)bh * NN * DHEAD;
    const unsigned short* km = k  + (size_t)bh * NN * DHEAD;
    const unsigned short* vm = vt + (size_t)bh * DHEAD * NN;
    const int* img = image_ids + (size_t)b * NN;
    const unsigned char* pd = pad + (size_t)b * NN;
    unsigned short* Pw = Plds + wid * 512;

    int r  = lane & 15;
    int kbp = (lane >> 4) * 8;               // fragment K packing offset
    int c  = lane & 15;
    int m0 = (lane >> 4) * 8;                // C-layout row base

    // Q fragments (rows qb*16 .. +15, two K=32 halves of D=64)
    Frag16 qf[2];
    #pragma unroll
    for (int dh = 0; dh < 2; dh++) {
        const unsigned short* p = qm + (size_t)(qb * 16 + r) * DHEAD + dh * 32 + kbp;
        qf[dh].u[0] = *(const v4u*)p;
        qf[dh].u[1] = *(const v4u*)(p + 16);
    }
    int imgq[8];
    #pragma unroll
    for (int i = 0; i < 8; i++) imgq[i] = img[qb * 16 + m0 + i];

    float mrow[8], lrow[8];
    v8f O[4] = {};
    #pragma unroll
    for (int i = 0; i < 8; i++) { mrow[i] = -3.0e38f; lrow[i] = 0.0f; }

    for (int j = 0; j < NN / 32; j++) {
        int kb32 = j * 32;
        // ---- S = Q x K^T : 16 x 32 scores (two 16-col tiles)
        AccF s0, s1; s0.f = (v8f){}; s1.f = (v8f){};
        #pragma unroll
        for (int s = 0; s < 2; s++) {
            Frag16 kf0, kf1;
            const unsigned short* p0 = km + (size_t)(kb32 + s * 16 + c) * DHEAD + kbp;
            kf0.u[0] = *(const v4u*)p0;        kf0.u[1] = *(const v4u*)(p0 + 16);
            kf1.u[0] = *(const v4u*)(p0 + 32); kf1.u[1] = *(const v4u*)(p0 + 48);
            if (s == 0) { s0.f = wmma_bf16(qf[0], kf0, s0.f); s0.f = wmma_bf16(qf[1], kf1, s0.f); }
            else        { s1.f = wmma_bf16(qf[0], kf0, s1.f); s1.f = wmma_bf16(qf[1], kf1, s1.f); }
        }
        // ---- block-diagonal + key-pad mask
        int key0 = kb32 + c, key1 = key0 + 16;
        int ik0 = img[key0], ik1 = img[key1];
        bool ok0 = (pd[key0] == 0), ok1 = (pd[key1] == 0);
        #pragma unroll
        for (int i = 0; i < 8; i++) {
            if (imgq[i] != ik0 || !ok0) s0.e[i] = -3.4e38f;
            if (imgq[i] != ik1 || !ok1) s1.e[i] = -3.4e38f;
        }
        // ---- online softmax (row = 16 lanes of a half-wave)
        float alpha[8];
        #pragma unroll
        for (int i = 0; i < 8; i++) {
            float mx = fmaxf(s0.e[i], s1.e[i]);
            #pragma unroll
            for (int d = 1; d < 16; d <<= 1) mx = fmaxf(mx, __shfl_xor(mx, d, 32));
            float nm = fmaxf(mrow[i], mx);
            alpha[i] = __expf(mrow[i] - nm);
            mrow[i] = nm;
            s0.e[i] = __expf(s0.e[i] - nm);
            s1.e[i] = __expf(s1.e[i] - nm);
            float ps = s0.e[i] + s1.e[i];
            #pragma unroll
            for (int d = 1; d < 16; d <<= 1) ps += __shfl_xor(ps, d, 32);
            lrow[i] = lrow[i] * alpha[i] + ps;
        }
        #pragma unroll
        for (int t4 = 0; t4 < 4; t4++) {
            AccF ot; ot.f = O[t4];
            #pragma unroll
            for (int i = 0; i < 8; i++) ot.e[i] *= alpha[i];
            O[t4] = ot.f;
        }
        // ---- P: C-layout -> A-layout via LDS bounce
        #pragma unroll
        for (int i = 0; i < 8; i++) {
            Pw[(m0 + i) * 32 + c]      = f2bf(s0.e[i]);
            Pw[(m0 + i) * 32 + 16 + c] = f2bf(s1.e[i]);
        }
        asm volatile("s_wait_dscnt 0x0" ::: "memory");
        Frag16 pf;
        pf.u[0] = *(const v4u*)&Pw[r * 32 + kbp];
        pf.u[1] = *(const v4u*)&Pw[r * 32 + kbp + 16];
        // ---- O += P x V  (V transposed -> contiguous b128 fragment loads)
        #pragma unroll
        for (int t4 = 0; t4 < 4; t4++) {
            Frag16 vf;
            const unsigned short* p = vm + (size_t)(t4 * 16 + c) * NN + kb32 + kbp;
            vf.u[0] = *(const v4u*)p;
            vf.u[1] = *(const v4u*)(p + 16);
            O[t4] = wmma_bf16(pf, vf, O[t4]);
        }
    }
    // ---- normalize and write concat output [tok, h*64+d]
    float inv[8];
    #pragma unroll
    for (int i = 0; i < 8; i++) inv[i] = (lrow[i] > 0.0f) ? 1.0f / lrow[i] : 0.0f;
    #pragma unroll
    for (int t4 = 0; t4 < 4; t4++) {
        AccF ot; ot.f = O[t4];
        #pragma unroll
        for (int i = 0; i < 8; i++) {
            size_t tok = (size_t)b * NN + qb * 16 + m0 + i;
            o[tok * INNER + hh * DHEAD + t4 * 16 + c] = f2bf(ot.e[i] * inv[i]);
        }
    }
}

// =====================================================================
// Host-side orchestration
// =====================================================================
extern "C" void kernel_launch(void* const* d_in, const int* in_sizes, int n_in,
                              void* d_out, int out_size, void* d_ws, size_t ws_size,
                              hipStream_t stream) {
    const float* patches   = (const float*)d_in[0];
    const int*   pos_h     = (const int*)d_in[1];
    const int*   pos_w     = (const int*)d_in[2];
    const int*   image_ids = (const int*)d_in[3];
    const unsigned char* pad = (const unsigned char*)d_in[4];
    const float* pn_n      = (const float*)d_in[5];
    const float* pn_mean   = (const float*)d_in[6];
    const float* pn_m2     = (const float*)d_in[7];
    const float* peh       = (const float*)d_in[8];
    const float* pew       = (const float*)d_in[9];
    const float* pe_w      = (const float*)d_in[10];
    const float* pe_b      = (const float*)d_in[11];
    const float* pe_ln_g   = (const float*)d_in[12];
    const float* attn_ln_g = (const float*)d_in[13];
    const float* Wq        = (const float*)d_in[14];
    const float* Wkv       = (const float*)d_in[15];
    const float* Wo        = (const float*)d_in[16];
    const float* qn_g      = (const float*)d_in[17];
    const float* kn_g      = (const float*)d_in[18];
    const float* ff_ln_g   = (const float*)d_in[19];
    const float* ff_w1     = (const float*)d_in[20];
    const float* ff_b1     = (const float*)d_in[21];
    const float* ff_w2     = (const float*)d_in[22];
    const float* ff_b2     = (const float*)d_in[23];
    const float* fin_ln_g  = (const float*)d_in[24];

    // ---- workspace layout ----
    char* ws = (char*)d_ws;
    size_t off = 0;
    float* x = (float*)(ws + off);           off += (size_t)MTOK * DIM * 4;      // residual
    float* tmp = (float*)(ws + off);                                             // alias w/ ff1
    unsigned short* ff1 = (unsigned short*)(ws + off); off += (size_t)MTOK * MLP * 2;
    unsigned short* h = (unsigned short*)(ws + off);   off += (size_t)MTOK * DIM * 2;
    unsigned short* qraw = (unsigned short*)(ws + off); off += (size_t)MTOK * INNER * 2;
    unsigned short* a0 = (unsigned short*)(ws + off);                             // alias w/ kvraw
    unsigned short* kvraw = (unsigned short*)(ws + off); off += (size_t)MTOK * 2 * INNER * 2;
    unsigned short* qb = (unsigned short*)(ws + off);  off += (size_t)MTOK * INNER * 2;
    unsigned short* kb = (unsigned short*)(ws + off);  off += (size_t)MTOK * INNER * 2;
    unsigned short* vtb = (unsigned short*)(ws + off); off += (size_t)MTOK * INNER * 2;
    unsigned short* ob = (unsigned short*)(ws + off);  off += (size_t)MTOK * INNER * 2;

    dim3 blk(256);
    // ---- PatchNorm + pos embed -> a0 (bf16) ----
    patchnorm_kernel<<<MTOK, blk, 0, stream>>>(patches, pos_h, pos_w, pad,
        pn_n, pn_mean, pn_m2, peh, pew, a0);
    // ---- patch embedding: tmp = a0 @ pe_w + pe_b ; x = LN(tmp) ----
    gemm_kernel<false,false,false><<<dim3(DIM/128, MTOK/128), blk, 0, stream>>>(
        a0, pe_w, pe_b, tmp, nullptr, nullptr, DIM, PDIM);
    ln_kernel<false><<<MTOK, blk, 0, stream>>>(tmp, pe_ln_g, x, nullptr);

    for (int l = 0; l < DEPTH; l++) {
        const float* wq = Wq + (size_t)l * DIM * INNER;
        const float* wkv = Wkv + (size_t)l * DIM * 2 * INNER;
        const float* wo = Wo + (size_t)l * INNER * DIM;
        // h = LN(x)
        ln_kernel<true><<<MTOK, blk, 0, stream>>>(x, attn_ln_g + l * DIM, nullptr, h);
        // qraw = h @ Wq ; kvraw = h @ Wkv
        gemm_kernel<true,false,false><<<dim3(INNER/128, MTOK/128), blk, 0, stream>>>(
            h, wq, nullptr, nullptr, qraw, nullptr, INNER, DIM);
        gemm_kernel<true,false,false><<<dim3(2*INNER/128, MTOK/128), blk, 0, stream>>>(
            h, wkv, nullptr, nullptr, kvraw, nullptr, 2 * INNER, DIM);
        // QK rmsnorm + V transpose
        qkv_transform_kernel<<<(BB*HEADS*NN)/8, blk, 0, stream>>>(
            qraw, kvraw, qn_g + (size_t)l * HEADS * DHEAD,
            kn_g + (size_t)l * HEADS * DHEAD, qb, kb, vtb);
        // flash attention -> ob
        attn_kernel<<<(BB*HEADS*(NN/16))/8, blk, 0, stream>>>(
            qb, kb, vtb, image_ids, pad, ob);
        // x = ob @ Wo + x
        gemm_kernel<false,false,true><<<dim3(DIM/128, MTOK/128), blk, 0, stream>>>(
            ob, wo, nullptr, x, nullptr, x, DIM, INNER);
        // h = LN(x) ; ff1 = gelu(h @ w1 + b1) ; x = ff1 @ w2 + b2 + x
        ln_kernel<true><<<MTOK, blk, 0, stream>>>(x, ff_ln_g + l * DIM, nullptr, h);
        gemm_kernel<true,true,false><<<dim3(MLP/128, MTOK/128), blk, 0, stream>>>(
            h, ff_w1 + (size_t)l * DIM * MLP, ff_b1 + (size_t)l * MLP,
            nullptr, ff1, nullptr, MLP, DIM);
        gemm_kernel<false,false,true><<<dim3(DIM/128, MTOK/128), blk, 0, stream>>>(
            ff1, ff_w2 + (size_t)l * MLP * DIM, ff_b2 + (size_t)l * DIM,
            x, nullptr, x, DIM, MLP);
    }
    // ---- final LN -> d_out (f32) ----
    ln_kernel<false><<<MTOK, blk, 0, stream>>>(x, fin_ln_g, (float*)d_out, nullptr);
}
